// SentimentDisentangler_75282186764934
// MI455X (gfx1250) — matrix operand order repaired and checked by reading, not compiled, not run on hardware
//
#include <hip/hip_runtime.h>
#include <hip/hip_bf16.h>
#include <cstddef>
#include <cstdint>

// ---------------------------------------------------------------------------
// SentimentDisentangler on MI455X (gfx1250, wave32, WMMA bf16 / f32-accum)
//
// Pipeline (attention is identity: softmax over a size-1 axis == 1.0):
//   K1/K2: pre-swizzle W_ih[0:256,:] and W_hh into WMMA B-fragment layout (bf16)
//   K3:    cvec[dir] = fixed_vec @ W_ih[256:384,:] + b   (constant gate bias)
//   K4:    gates[dir,s,b,:] = emb[tok(b,s)] @ W_ih_top + cvec   (WMMA GEMM)
//   K5:    persistent recurrence. Batch rows are independent across the scan,
//          so 8 workgroups = 2 dirs x 4 row-blocks of 16. W_hh fragments live
//          in LDS; h is double-buffered in LDS in exact A-fragment layout;
//          cell state c lives in registers; the next step's 32 KB gate tile is
//          staged into LDS by the Tensor Data Mover (tensor_load_to_lds,
//          s_wait_tensorcnt) overlapped with the current step's WMMA chain.
//   K6:    head: sigmoid(dense(concat(h_f,h_b))) -> sigmoid(dense) -> out[64]
//
// Workspace: ~68 MB (gates 64MB + swizzled weights 0.75MB + cvec/hfin).
// ---------------------------------------------------------------------------

#define B_  64
#define S_  256
#define E_  256
#define H4_ 512   // 4*H
#define H_  128

typedef __bf16 bf16_t;
typedef bf16_t v16bf __attribute__((ext_vector_type(16)));
typedef float  v8f   __attribute__((ext_vector_type(8)));
typedef unsigned u32x4 __attribute__((ext_vector_type(4)));
typedef int      i32x4 __attribute__((ext_vector_type(4)));
typedef int      i32x8 __attribute__((ext_vector_type(8)));

union Frag { v16bf v; unsigned u[8]; };

#if __has_builtin(__builtin_amdgcn_tensor_load_to_lds) && __has_builtin(__builtin_amdgcn_s_wait_tensorcnt)
#define SD_HAS_TDM 1
#else
#define SD_HAS_TDM 0
#endif

__device__ __forceinline__ unsigned pk_bf16(float a, float b) {
    unsigned short x = __builtin_bit_cast(unsigned short, (bf16_t)a);
    unsigned short y = __builtin_bit_cast(unsigned short, (bf16_t)b);
    return (unsigned)x | ((unsigned)y << 16);
}

__device__ __forceinline__ float sigm(float x) { return 1.0f / (1.0f + __expf(-x)); }
__device__ __forceinline__ float ftanh(float x) {
    float e = __expf(2.0f * x);
    return 1.0f - 2.0f / (e + 1.0f);
}

#if SD_HAS_TDM
// Issue one TDM load of a contiguous 16x512 f32 tile into LDS.
// D# bit layout per CDNA5 ISA ch.8 (group0: count/lds_addr/global_addr/type=2;
// group1: data_size=4B, tensor_dim0=512, tensor_dim1=16, tile_dim0=512,
// tile_dim1=16, tensor_dim0_stride=512). Uniform operands -> SGPR groups.
__device__ __forceinline__ void tdm_load_tile(const float* gsrc, unsigned lds_off) {
    unsigned long long ga = (unsigned long long)(uintptr_t)gsrc;
    u32x4 g0 = {1u,                                   // count=1 (valid user D#)
                lds_off,                              // lds_addr (bytes)
                (unsigned)(ga & 0xffffffffull),       // global_addr[31:0]
                (unsigned)(ga >> 32) | (2u << 30)};   // global_addr[56:32] | type=2
    i32x8 g1 = {(int)(2u << 16),       // wg_mask=0, data_size=2 (4 bytes)
                (int)(512u << 16),     // tensor_dim0[15:0]=512 (bits 63:48)
                (int)(16u << 16),      // tensor_dim0 hi=0 | tensor_dim1[15:0]=16
                (int)(512u << 16),     // tensor_dim1 hi=0 | tile_dim0=512
                16,                    // tile_dim1=16, tile_dim2=0
                512,                   // tensor_dim0_stride[31:0]=512
                0, 0};                 // stride hi / tensor_dim1_stride=0
    i32x4 zg = {0, 0, 0, 0};
#if defined(__clang_major__) && (__clang_major__ >= 23)
    i32x8 zg8 = {0, 0, 0, 0, 0, 0, 0, 0};
    __builtin_amdgcn_tensor_load_to_lds(g0, g1, zg, zg, zg8, 0);
#else
    __builtin_amdgcn_tensor_load_to_lds(g0, g1, zg, zg, 0);
#endif
}
#endif

// --- K1: swizzle W_ih rows [0,256) into B-fragment layout ------------------
// layout: [dir][kc(8)][nt(32)][lane(32)][j(8)] uints; dword j of lane l holds
// K = (j>=4 ? 16:0) + (l>=16 ? 8:0) + 2(j&3) (+1), column N = nt*16 + (l&15).
__global__ __launch_bounds__(256) void sd_swz_ih(const float* __restrict__ Wf,
                                                 const float* __restrict__ Wb,
                                                 unsigned* __restrict__ out) {
    int i = blockIdx.x * 256 + threadIdx.x;
    if (i >= 2 * 65536) return;
    int dir = i >> 16, r = i & 65535;
    int j = r & 7, lane = (r >> 3) & 31, nt = (r >> 8) & 31, kc = (r >> 13) & 7;
    int koff = (lane >> 4) * 8, n = lane & 15;
    int k = kc * 32 + ((j >> 2) << 4) + koff + 2 * (j & 3);
    int col = nt * 16 + n;
    const float* W = dir ? Wb : Wf;
    out[i] = pk_bf16(W[(size_t)k * H4_ + col], W[(size_t)(k + 1) * H4_ + col]);
}

// --- K2: swizzle W_hh (128x512) into B-fragment layout ---------------------
// layout: [dir][kc(4)][nt(32)][lane(32)][j(8)] uints
__global__ __launch_bounds__(256) void sd_swz_hh(const float* __restrict__ Wf,
                                                 const float* __restrict__ Wb,
                                                 unsigned* __restrict__ out) {
    int i = blockIdx.x * 256 + threadIdx.x;
    if (i >= 2 * 32768) return;
    int dir = i >> 15, r = i & 32767;
    int j = r & 7, lane = (r >> 3) & 31, nt = (r >> 8) & 31, kc = (r >> 13) & 3;
    int koff = (lane >> 4) * 8, n = lane & 15;
    int k = kc * 32 + ((j >> 2) << 4) + koff + 2 * (j & 3);
    int col = nt * 16 + n;
    const float* W = dir ? Wb : Wf;
    out[i] = pk_bf16(W[(size_t)k * H4_ + col], W[(size_t)(k + 1) * H4_ + col]);
}

// --- K3: cvec[dir][n] = b[n] + sum_k fixed[k] * W_ih[256+k][n] -------------
__global__ void sd_cvec(const float* __restrict__ fixedv,
                        const float* __restrict__ Wf, const float* __restrict__ Wb,
                        const float* __restrict__ bfv, const float* __restrict__ bbv,
                        float* __restrict__ cvec) {
    int dir = blockIdx.x, n = threadIdx.x;  // 2 x 512
    const float* W = dir ? Wb : Wf;
    const float* bb = dir ? bbv : bfv;
    float a = bb[n];
    for (int k = 0; k < 128; ++k) a = fmaf(fixedv[k], W[(size_t)(E_ + k) * H4_ + n], a);
    cvec[dir * H4_ + n] = a;
}

// --- K4: gates[dir][s][b][:] = emb[tok(b,s)] @ W_ih_top + cvec -------------
// one workgroup per (s, dir): [64,256] x [256,512] with v_wmma_f32_16x16x32_bf16
__global__ __launch_bounds__(256) void sd_gates(const int* __restrict__ tokens,
                                                const float* __restrict__ emb,
                                                const unsigned* __restrict__ ihsw,
                                                const float* __restrict__ cvec,
                                                float* __restrict__ gates) {
    __shared__ bf16_t Abuf[B_ * E_];  // 32 KB, gathered embeddings in bf16
    const int s = blockIdx.x, dir = blockIdx.y, t = threadIdx.x;
    {
        const int row = t >> 2, seg = t & 3;
        const int tok = tokens[row * S_ + s];
        const float4* src = (const float4*)(emb + (size_t)tok * E_ + seg * 64);
        bf16_t* dst = &Abuf[row * E_ + seg * 64];
#pragma unroll
        for (int i = 0; i < 16; ++i) {
            float4 v = src[i];
            dst[i * 4 + 0] = (bf16_t)v.x; dst[i * 4 + 1] = (bf16_t)v.y;
            dst[i * 4 + 2] = (bf16_t)v.z; dst[i * 4 + 3] = (bf16_t)v.w;
        }
    }
    __syncthreads();

    const int w = t >> 5, lane = t & 31;
    const int mt = w & 3, ntBase = (w >> 2) * 16;
    const int n = lane & 15, half = lane >> 4;
    const int m0 = mt * 16 + half * 8;
    const unsigned* Bsw = ihsw + (size_t)dir * 65536;
    float* gout = gates + ((size_t)(dir * S_ + s)) * B_ * H4_;
    const unsigned* Au = (const unsigned*)Abuf;
    const int arow = mt * 16 + n;  // A row for this lane (m = lane&15)

    for (int ti = 0; ti < 16; ++ti) {
        const int nt = ntBase + ti;
        const float cv = cvec[dir * H4_ + nt * 16 + n];
        v8f acc = {cv, cv, cv, cv, cv, cv, cv, cv};
#pragma unroll
        for (int kc = 0; kc < 8; ++kc) {
            Frag A, Bg;
#pragma unroll
            for (int j = 0; j < 8; ++j) {
                const int kb = ((j >= 4) ? 16 : 0) + (half ? 8 : 0) + ((j & 3) << 1);
                A.u[j] = Au[(arow * E_ + kc * 32 + kb) >> 1];
            }
            const uint4* bp = (const uint4*)(Bsw + ((kc * 32 + nt) * 32 + lane) * 8);
            uint4 b0 = bp[0], b1 = bp[1];
            Bg.u[0] = b0.x; Bg.u[1] = b0.y; Bg.u[2] = b0.z; Bg.u[3] = b0.w;
            Bg.u[4] = b1.x; Bg.u[5] = b1.y; Bg.u[6] = b1.z; Bg.u[7] = b1.w;
            acc = __builtin_amdgcn_wmma_f32_16x16x32_bf16(false, A.v, false, Bg.v,
                                                          (short)0, acc, false, false);
        }
        const int ncol = nt * 16 + n;
#pragma unroll
        for (int v = 0; v < 8; ++v) gout[(size_t)(m0 + v) * H4_ + ncol] = acc[v];
    }
}

// --- K5: persistent bidirectional LSTM recurrence --------------------------
// 8 workgroups: blockIdx.x = dir*4 + row-block (16 batch rows each; batch rows
// are independent through the scan). Wave w owns hidden units [w*16,(w+1)*16)
// and all four gate quadrants for them -> cell state in registers, no LDS
// round-trip of pre-activations. TDM stages the next step's gate tile.
__global__ __launch_bounds__(256) void sd_recur(const float* __restrict__ gates,
                                                const unsigned* __restrict__ hhsw,
                                                float* __restrict__ hfin) {
    __shared__ unsigned Bfrag[4 * 32 * 32 * 8];   // 128 KB: W_hh fragments
    __shared__ bf16_t hAbuf[2][4 * 32 * 16];      // 2 x 4 KB: h in A-frag layout
    __shared__ float Gs[2][16 * H4_];             // 2 x 32 KB: staged gate tiles
    const int dir = blockIdx.x >> 2, rb = blockIdx.x & 3;
    const int t = threadIdx.x;
    {   // stage swizzled W_hh into LDS (reused 256 steps)
        const uint4* src = (const uint4*)(hhsw + (size_t)dir * 32768);
        uint4* dst = (uint4*)Bfrag;
        for (int i = t; i < 8192; i += 256) dst[i] = src[i];
        uint4 z; z.x = z.y = z.z = z.w = 0u;
        uint4* dz = (uint4*)&hAbuf[0][0];
        for (int i = t; i < 256; i += 256) dz[i] = z;  // h0 = 0
    }
    const float* Gbase = gates + ((size_t)dir * S_) * B_ * H4_ + (size_t)rb * 16 * H4_;
#if SD_HAS_TDM
    if (t < 32) {  // one wave issues the TDM for step 0
        const int s0 = dir ? (S_ - 1) : 0;
        tdm_load_tile(Gbase + (size_t)s0 * B_ * H4_,
                      (unsigned)(uintptr_t)&Gs[0][0]);
    }
#endif
    __syncthreads();

    const int w = t >> 5, lane = t & 31;
    const int n = lane & 15, half = lane >> 4;
    const int m0 = half * 8;       // row within this 16-row block
    const int u = w * 16 + n;      // hidden unit owned by this lane column

    float creg[8];
#pragma unroll
    for (int v = 0; v < 8; ++v) creg[v] = 0.0f;

    int p = 0;
    for (int step = 0; step < S_; ++step) {
        const int cur = step & 1;
#if SD_HAS_TDM
        if (t < 32) __builtin_amdgcn_s_wait_tensorcnt(0);  // Gs[cur] landed
        __syncthreads();                                   // publish to all waves
        if (step + 1 < S_ && t < 32) {                     // overlap next tile
            const int sn = dir ? (S_ - 2 - step) : step + 1;
            tdm_load_tile(Gbase + (size_t)sn * B_ * H4_,
                          (unsigned)(uintptr_t)&Gs[cur ^ 1][0]);
        }
#else
        {   // fallback: cooperative global->LDS stage of this step's tile
            const int sg = dir ? (S_ - 1 - step) : step;
            const float* gsrc = Gbase + (size_t)sg * B_ * H4_;
            float* gd = &Gs[cur][0];
            for (int i = t; i < 16 * H4_; i += 256) gd[i] = gsrc[i];
        }
        __syncthreads();
#endif
        // A fragments of h_{t-1} (4 K-chunks), shared by all 4 gate quadrants
        Frag A[4];
        const unsigned* hp = (const unsigned*)&hAbuf[p][0];
#pragma unroll
        for (int kc = 0; kc < 4; ++kc) {
            const unsigned* ap = hp + (kc * 32 + lane) * 8;
#pragma unroll
            for (int j = 0; j < 8; ++j) A[kc].u[j] = ap[j];
        }
        bf16_t* hw = &hAbuf[1 - p][0];
        const float* Gc = &Gs[cur][0];

        v8f acc[4];
#pragma unroll
        for (int q = 0; q < 4; ++q) {   // gate quadrants i,f,g,o
            const float* gp = Gc + q * H_ + w * 16 + n;
#pragma unroll
            for (int v = 0; v < 8; ++v) acc[q][v] = gp[(m0 + v) * H4_];
        }
#pragma unroll
        for (int q = 0; q < 4; ++q) {
            const int nt = q * 8 + w;
#pragma unroll
            for (int kc = 0; kc < 4; ++kc) {
                Frag Bg;
                const uint4* bp = (const uint4*)(Bfrag + ((kc * 32 + nt) * 32 + lane) * 8);
                uint4 b0 = bp[0], b1 = bp[1];
                Bg.u[0] = b0.x; Bg.u[1] = b0.y; Bg.u[2] = b0.z; Bg.u[3] = b0.w;
                Bg.u[4] = b1.x; Bg.u[5] = b1.y; Bg.u[6] = b1.z; Bg.u[7] = b1.w;
                acc[q] = __builtin_amdgcn_wmma_f32_16x16x32_bf16(false, A[kc].v, false, Bg.v,
                                                                 (short)0, acc[q], false, false);
            }
        }
        // LSTM cell update; write h_t into the other hA buffer (A-frag layout)
#pragma unroll
        for (int v = 0; v < 8; ++v) {
            const int b = m0 + v;   // row within block
            float c = creg[v];
            c = sigm(acc[1][v] + 1.0f) * c + sigm(acc[0][v]) * ftanh(acc[2][v]);
            const float h = sigm(acc[3][v]) * ftanh(c);
            creg[v] = c;
            const int krel = u & 31, kc2 = u >> 5;
            const int hi = krel >> 4, k15 = krel & 15, hf2 = k15 >> 3;
            const int jj = hi * 4 + ((k15 & 7) >> 1), lo = krel & 1;
            const int lane2 = (b & 15) + 16 * hf2;
            hw[(kc2 * 32 + lane2) * 16 + jj * 2 + lo] = (bf16_t)h;
            if (step == S_ - 1)
                hfin[((size_t)dir * B_ + rb * 16 + b) * H_ + u] = h;
        }
        __syncthreads();
        p ^= 1;
    }
}

// --- K6: head: sigmoid(concat(h_f,h_b) @ Wd + bd) @ Wo + bo ----------------
__global__ __launch_bounds__(256) void sd_head(const float* __restrict__ hfin,
                                               const float* __restrict__ Wd,
                                               const float* __restrict__ bd,
                                               const float* __restrict__ Wo,
                                               const float* __restrict__ bo,
                                               float* __restrict__ out) {
    __shared__ float L[B_ * 256];   // 64 KB
    __shared__ float D1[B_ * 128];  // 32 KB
    const int t = threadIdx.x;
    for (int i = t; i < B_ * 256; i += 256) {
        int b = i >> 8, k = i & 255;
        L[i] = (k < 128) ? hfin[b * H_ + k] : hfin[B_ * H_ + b * H_ + (k - 128)];
    }
    __syncthreads();
    for (int i = t; i < B_ * 128; i += 256) {
        int b = i >> 7, j = i & 127;
        float a = bd[j];
        for (int k = 0; k < 256; ++k) a = fmaf(L[b * 256 + k], Wd[k * 128 + j], a);
        D1[i] = sigm(a);
    }
    __syncthreads();
    if (t < B_) {
        float a = bo[0];
        for (int j = 0; j < 128; ++j) a = fmaf(D1[t * 128 + j], Wo[j], a);
        out[t] = sigm(a);
    }
}

// ---------------------------------------------------------------------------
extern "C" void kernel_launch(void* const* d_in, const int* in_sizes, int n_in,
                              void* d_out, int out_size, void* d_ws, size_t ws_size,
                              hipStream_t stream) {
    (void)in_sizes; (void)n_in; (void)out_size; (void)ws_size;
    const int*   tokens = (const int*)  d_in[0];
    const float* emb    = (const float*)d_in[1];
    const float* fixedv = (const float*)d_in[2];
    // d_in[3] attn_w, d_in[4] attn_b: dead (softmax over size-1 axis == 1)
    const float* Wf_ih = (const float*)d_in[5];
    const float* Wf_hh = (const float*)d_in[6];
    const float* bfv   = (const float*)d_in[7];
    const float* Wb_ih = (const float*)d_in[8];
    const float* Wb_hh = (const float*)d_in[9];
    const float* bbv   = (const float*)d_in[10];
    const float* Wd    = (const float*)d_in[11];
    const float* bd    = (const float*)d_in[12];
    const float* Wo    = (const float*)d_in[13];
    const float* bo    = (const float*)d_in[14];
    float* out = (float*)d_out;

    // workspace layout (needs ~68 MB)
    char* ws = (char*)d_ws;
    float*    g_gates = (float*)ws;                               // 2*256*64*512 f32 = 64 MB
    unsigned* ihsw    = (unsigned*)(ws + 67108864);               // 512 KB
    unsigned* hhsw    = (unsigned*)(ws + 67108864 + 524288);      // 256 KB
    float*    cvec    = (float*)(ws + 67108864 + 524288 + 262144);          // 4 KB
    float*    hfin    = (float*)(ws + 67108864 + 524288 + 262144 + 4096);   // 64 KB

    sd_swz_ih<<<512, 256, 0, stream>>>(Wf_ih, Wb_ih, ihsw);
    sd_swz_hh<<<256, 256, 0, stream>>>(Wf_hh, Wb_hh, hhsw);
    sd_cvec  <<<2, 512, 0, stream>>>(fixedv, Wf_ih, Wb_ih, bfv, bbv, cvec);
    sd_gates <<<dim3(S_, 2), 256, 0, stream>>>(tokens, emb, ihsw, cvec, g_gates);
    sd_recur <<<8, 256, 0, stream>>>(g_gates, hhsw, hfin);
    sd_head  <<<1, 256, 0, stream>>>(hfin, Wd, bd, Wo, bo, out);
}
